// QSAPatchSampler_63625645523283
// MI455X (gfx1250) — matrix operand order repaired and checked by reading, not compiled, not run on hardware
//
#include <hip/hip_runtime.h>
#include <hip/hip_bf16.h>

// ---------------------------------------------------------------------------
// QSAPatchSampler for MI455X (gfx1250, wave32)
// WMMA f32_16x16x32_f16 for all GEMMs + TDM (tensor_load_to_lds) staging of
// the B-operand column stream in the dominant entropy pass.
// B=4, C=64, H=W=64 -> N=4096 patches, P=K=256
// ---------------------------------------------------------------------------

typedef __attribute__((ext_vector_type(16))) _Float16 v16h;
typedef __attribute__((ext_vector_type(8)))  _Float16 v8h;
typedef __attribute__((ext_vector_type(8)))  float    v8f;

typedef unsigned int uint32x4 __attribute__((ext_vector_type(4)));
typedef int          int32x8  __attribute__((ext_vector_type(8)));
typedef int          int32x4  __attribute__((ext_vector_type(4)));

#define NB   4
#define NC   64
#define NN   4096
#define NP   256
#define NK   256
#define JT   64   // columns staged per TDM chunk (8KB)
#define NIT  (NN / JT)

static __device__ __forceinline__ v8f wmma16(v16h a, v16h b, v8f c) {
  // D = A(16x32,f16) x B(32x16,f16) + C(16x16,f32)
  return __builtin_amdgcn_wmma_f32_16x16x32_f16(
      /*neg_a=*/false, a, /*neg_b=*/false, b,
      /*c_mod=*/(short)0, c, /*reuse_a=*/false, /*reuse_b=*/false);
}

static __device__ __forceinline__ v16h cat8(v8h lo, v8h hi) {
  return __builtin_shufflevector(lo, hi, 0, 1, 2, 3, 4, 5, 6, 7,
                                          8, 9, 10, 11, 12, 13, 14, 15);
}

// Issue a TDM load of `JT` consecutive pb rows (each NC f16 = 128B) into LDS.
// D# per CDNA5 ISA ch.8: group0 = {count/lds_addr/global_addr/type},
// group1 = {data_size, tensor/tile dims, stride}; 2-D tile so groups 2/3 = 0.
static __device__ __forceinline__ void tdm_load_rows(const _Float16* gsrc,
                                                     unsigned int lds_addr) {
  unsigned long long ga = (unsigned long long)(unsigned long long)(size_t)gsrc;
  uint32x4 g0;
  g0.x = 1u;                                   // count = 1 (valid descriptor)
  g0.y = lds_addr;                             // LDS byte address
  g0.z = (unsigned int)ga;                     // global_addr[31:0]
  g0.w = (unsigned int)((ga >> 32) & 0x1FFFFFFu) | (2u << 30);  // type=2
  int32x8 g1;
  g1[0] = (1 << 16);                 // data_size = 1 (2-byte elements)
  g1[1] = (NC << 16);                // tensor_dim0[15:0] = 64 (bits 63:48)
  g1[2] = ((NN & 0xFFFF) << 16);     // tensor_dim0 hi=0, tensor_dim1 lo=4096
  g1[3] = (NC << 16);                // tensor_dim1 hi=0, tile_dim0 = 64
  g1[4] = JT;                        // tile_dim1 = 64 rows, tile_dim2 = 0
  g1[5] = NC;                        // tensor_dim0_stride[31:0] = 64
  g1[6] = 0;                         // stride hi / tensor_dim1_stride lo
  g1[7] = 0;
  int32x4 z4 = {0, 0, 0, 0};
#if __clang_major__ >= 23
  int32x8 z8 = {0, 0, 0, 0, 0, 0, 0, 0};
  __builtin_amdgcn_tensor_load_to_lds(g0, g1, z4, z4, z8, 0);
#else
  __builtin_amdgcn_tensor_load_to_lds(g0, g1, z4, z4, 0);
#endif
}

// ---------------------------------------------------------------------------
// K0a: patches f16 row-major [B][N][C] + per-row sumsq + per-batch max sumsq
// ---------------------------------------------------------------------------
__global__ __launch_bounds__(256) void k_prep_rows(
    const float* __restrict__ x, _Float16* __restrict__ pb,
    float* __restrict__ norm2, int* __restrict__ bmax) {
  int row = blockIdx.x * blockDim.x + threadIdx.x;   // 0 .. B*N-1
  if (row >= NB * NN) return;
  int b = row >> 12, n = row & (NN - 1);
  const float* src = x + (size_t)b * NC * NN + n;
  _Float16 tmp[NC];
  float ss = 0.f;
#pragma unroll
  for (int c = 0; c < NC; ++c) {
    float v = src[(size_t)c * NN];
    ss = fmaf(v, v, ss);
    tmp[c] = (_Float16)v;
  }
  v8h* dst = (v8h*)(pb + (size_t)row * NC);
#pragma unroll
  for (int i = 0; i < 8; ++i) {
    v8h w;
#pragma unroll
    for (int e = 0; e < 8; ++e) w[e] = tmp[i * 8 + e];
    dst[i] = w;
  }
  norm2[row] = ss;
  atomicMax(bmax + b, __float_as_int(ss));  // ss >= 0, int-bit compare is OK
}

// K0b: straight f32 -> f16 cast (layer_out is already patches^T = [B][C][N])
__global__ __launch_bounds__(256) void k_cvt(
    const float* __restrict__ src, _Float16* __restrict__ dst, int n) {
  int i = blockIdx.x * blockDim.x + threadIdx.x;
  if (i < n) dst[i] = (_Float16)src[i];
}

// K0c: transposed f16 weights
__global__ __launch_bounds__(256) void k_wt(
    const float* __restrict__ w1, const float* __restrict__ w2,
    _Float16* __restrict__ w1t, _Float16* __restrict__ w2t) {
  int i = blockIdx.x * blockDim.x + threadIdx.x;
  if (i < NP * NC) {
    int p = i / NC, c = i % NC;
    w1t[i] = (_Float16)w1[c * NP + p];
  }
  if (i < NP * NP) {
    int po = i / NP, pi = i % NP;
    w2t[i] = (_Float16)w2[pi * NP + po];
  }
}

// ---------------------------------------------------------------------------
// K1: entropy pass. 8 waves/WG, each wave owns a 16-row block; the shared
// 64-column B stream is TDM double-buffered through LDS (16KB) so each WG
// reads the 512KB per-batch pb buffer once instead of 8x.
// Shift s_i = ||x_i||*max_j||x_j|| >= max_j d_ij, so no running max:
//   Z = sum e^(d-s), S = sum d*e^(d-s), ent = s + log Z - S/Z.
// ---------------------------------------------------------------------------
__global__ __launch_bounds__(256) void k_entropy(
    const _Float16* __restrict__ pb, const float* __restrict__ norm2,
    const int* __restrict__ bmax, float* __restrict__ ent,
    float2* __restrict__ sZ) {
  __shared__ _Float16 sB[2][JT * NC];  // 2 x 8KB column-chunk buffers

  int tid = threadIdx.x;
  int lane = tid & 31;
  int wv = tid >> 5;                         // 0..7
  int wave = blockIdx.x * 8 + wv;            // 0..1023
  int b = wave >> 8;
  int i0 = (wave & 255) << 4;
  int l16 = lane & 15, h = lane >> 4;

  const _Float16* pbB = pb + (size_t)b * NN * NC;

  // prologue: stage chunk 0 (wave 0 issues; TDM ignores EXEC)
  if (tid == 0) {
    tdm_load_rows(pbB, (unsigned int)(size_t)&sB[0][0]);
  }

  // A tiles: rows i0..i0+15, per-lane chunks in contiguous k order (k = 16h + s)
  const v8h* arow = (const v8h*)(pbB + (size_t)(i0 + l16) * NC);
  v16h A0 = cat8(arow[2 * h],     arow[2 * h + 1]);  // K = 0..31
  v16h A1 = cat8(arow[4 + 2 * h], arow[5 + 2 * h]);  // K = 32..63

  float Mn = sqrtf(__int_as_float(bmax[b]));
  float sv[8], Zr[8], Sr[8];
#pragma unroll
  for (int r = 0; r < 8; ++r) {
    sv[r] = sqrtf(norm2[b * NN + i0 + r + 8 * h]) * Mn;
    Zr[r] = 0.f;
    Sr[r] = 0.f;
  }

  for (int it = 0; it < NIT; ++it) {
    if (tid == 0) {
      if (it + 1 < NIT) {  // prefetch next chunk into the other buffer
        tdm_load_rows(pbB + (size_t)(it + 1) * JT * NC,
                      (unsigned int)(size_t)&sB[(it + 1) & 1][0]);
        __builtin_amdgcn_s_wait_tensorcnt(1);  // current chunk complete
      } else {
        __builtin_amdgcn_s_wait_tensorcnt(0);
      }
    }
    __syncthreads();  // publish LDS chunk to all 8 waves

    const _Float16* cb = &sB[it & 1][0];
#pragma unroll
    for (int jt = 0; jt < JT / 16; ++jt) {
      const _Float16* brow = cb + (size_t)(jt * 16 + l16) * NC;
      v16h B0 = *(const v16h*)(brow + 16 * h);       // K = 0..31 (ds_load)
      v16h B1 = *(const v16h*)(brow + 32 + 16 * h);  // K = 32..63
      v8f acc = {};
      acc = wmma16(A0, B0, acc);
      acc = wmma16(A1, B1, acc);
#pragma unroll
      for (int r = 0; r < 8; ++r) {
        float d = acc[r];
        float e = __expf(d - sv[r]);
        Zr[r] += e;
        Sr[r] = fmaf(d, e, Sr[r]);
      }
    }
    __syncthreads();  // all waves done with this buffer before TDM reuses it
  }

  // reduce across the 16 columns (half-wave)
#pragma unroll
  for (int m = 8; m; m >>= 1) {
#pragma unroll
    for (int r = 0; r < 8; ++r) {
      Zr[r] += __shfl_xor(Zr[r], m, 16);
      Sr[r] += __shfl_xor(Sr[r], m, 16);
    }
  }
  if (l16 == 0) {
#pragma unroll
    for (int r = 0; r < 8; ++r) {
      int row = b * NN + i0 + r + 8 * h;
      float z = Zr[r];
      ent[row] = sv[r] + logf(z) - Sr[r] / z;
      sZ[row] = make_float2(sv[r], z);
    }
  }
}

// ---------------------------------------------------------------------------
// K2: per-batch LDS bitonic sort of 4096 (ent, idx) pairs, tie-break on idx
// ---------------------------------------------------------------------------
__global__ __launch_bounds__(256) void k_topk(
    const float* __restrict__ ent, int* __restrict__ idx_out) {
  __shared__ float sk[NN];
  __shared__ int   si[NN];
  int b = blockIdx.x;
  for (int i = threadIdx.x; i < NN; i += 256) {
    sk[i] = ent[b * NN + i];
    si[i] = i;
  }
  __syncthreads();
  for (int k = 2; k <= NN; k <<= 1) {
    for (int j = k >> 1; j > 0; j >>= 1) {
      for (int t = threadIdx.x; t < NN / 2; t += 256) {
        int i = ((t & ~(j - 1)) << 1) | (t & (j - 1));
        int p = i | j;
        bool up = ((i & k) == 0);
        float ka = sk[i], kb = sk[p];
        int ia = si[i], ib = si[p];
        bool agtb = (ka > kb) || (ka == kb && ia > ib);
        if (agtb == up) {
          sk[i] = kb; sk[p] = ka;
          si[i] = ib; si[p] = ia;
        }
      }
      __syncthreads();
    }
  }
  if (threadIdx.x < NK) idx_out[b * NK + threadIdx.x] = si[threadIdx.x];
}

// ---------------------------------------------------------------------------
// K3: recompute dots for selected rows, emit attn_k = exp(d - s) / Z
// ---------------------------------------------------------------------------
__global__ __launch_bounds__(256) void k_attn(
    const _Float16* __restrict__ pb, const float2* __restrict__ sZ,
    const int* __restrict__ idx, float* __restrict__ attn) {
  int lane = threadIdx.x & 31;
  int wave = (blockIdx.x * blockDim.x + threadIdx.x) >> 5;  // 0..63
  int b = wave >> 4;
  int i0 = (wave & 15) << 4;
  int l16 = lane & 15, h = lane >> 4;

  const _Float16* pbB = pb + (size_t)b * NN * NC;
  int ra = idx[b * NK + i0 + l16];
  const v8h* arow = (const v8h*)(pbB + (size_t)ra * NC);
  v16h A0 = cat8(arow[2 * h],     arow[2 * h + 1]);
  v16h A1 = cat8(arow[4 + 2 * h], arow[5 + 2 * h]);

  float sv[8], iz[8];
#pragma unroll
  for (int r = 0; r < 8; ++r) {
    int sel = idx[b * NK + i0 + r + 8 * h];
    float2 t = sZ[b * NN + sel];
    sv[r] = t.x;
    iz[r] = 1.f / t.y;
  }

  for (int j0 = 0; j0 < NN; j0 += 16) {
    const _Float16* brow = pbB + (size_t)(j0 + l16) * NC;
    v16h B0 = *(const v16h*)(brow + 16 * h);
    v16h B1 = *(const v16h*)(brow + 32 + 16 * h);
    v8f acc = {};
    acc = wmma16(A0, B0, acc);
    acc = wmma16(A1, B1, acc);
#pragma unroll
    for (int r = 0; r < 8; ++r) {
      attn[((size_t)b * NK + i0 + r + 8 * h) * NN + j0 + l16] =
          __expf(acc[r] - sv[r]) * iz[r];
    }
  }
}

// ---------------------------------------------------------------------------
// K4: sampled = attn_k @ patches  ([256 x 4096] x [4096 x 64] per batch)
// ---------------------------------------------------------------------------
__global__ __launch_bounds__(256) void k_sampled(
    const float* __restrict__ attn, const _Float16* __restrict__ pbT,
    float* __restrict__ sampled) {
  int lane = threadIdx.x & 31;
  int wave = (blockIdx.x * blockDim.x + threadIdx.x) >> 5;  // 0..63
  int b = wave >> 4;
  int i0 = (wave & 15) << 4;
  int l16 = lane & 15, h = lane >> 4;

  v8f acc[4] = {{}, {}, {}, {}};
  const float* arow = attn + ((size_t)b * NK + i0 + l16) * NN;
  const _Float16* pT = pbT + (size_t)b * NC * NN;

  for (int k0 = 0; k0 < NN; k0 += 32) {
    v8f x0 = *(const v8f*)(arow + k0 + 16 * h);
    v8f x1 = *(const v8f*)(arow + k0 + 8 + 16 * h);
    v16h A;
#pragma unroll
    for (int e = 0; e < 8; ++e) {
      A[e] = (_Float16)x0[e];
      A[8 + e] = (_Float16)x1[e];
    }
#pragma unroll
    for (int ct = 0; ct < 4; ++ct) {
      const _Float16* brow = pT + (size_t)(ct * 16 + l16) * NN + k0 + 16 * h;
      v16h Bv = *(const v16h*)brow;
      acc[ct] = wmma16(A, Bv, acc[ct]);
    }
  }
#pragma unroll
  for (int ct = 0; ct < 4; ++ct)
#pragma unroll
    for (int r = 0; r < 8; ++r)
      sampled[((size_t)b * NK + i0 + r + 8 * h) * NC + ct * 16 + l16] =
          acc[ct][r];
}

// ---------------------------------------------------------------------------
// K5: h = relu(sampled@w1 + b1) @ w2 + b2 ; emb = h / max(||h||, eps)
// ---------------------------------------------------------------------------
__global__ __launch_bounds__(128) void k_mlp(
    const float* __restrict__ sampled, const _Float16* __restrict__ w1t,
    const _Float16* __restrict__ w2t, const float* __restrict__ b1,
    const float* __restrict__ b2, float* __restrict__ emb) {
  __shared__ _Float16 h1s[4][16 * NP];  // 4 waves * 8KB = 32KB
  int lane = threadIdx.x & 31, wv = threadIdx.x >> 5;
  int wave = blockIdx.x * 4 + wv;  // 0..63
  int b = wave >> 4;
  int i0 = (wave & 15) << 4;
  int l16 = lane & 15, h = lane >> 4;

  // A operands from sampled rows (f32 -> f16)
  const float* srow = sampled + ((size_t)b * NK + i0 + l16) * NC;
  v8f x0 = *(const v8f*)(srow + 16 * h);
  v8f x1 = *(const v8f*)(srow + 8 + 16 * h);
  v8f x2 = *(const v8f*)(srow + 32 + 16 * h);
  v8f x3 = *(const v8f*)(srow + 40 + 16 * h);
  v16h A0, A1;
#pragma unroll
  for (int e = 0; e < 8; ++e) {
    A0[e] = (_Float16)x0[e]; A0[8 + e] = (_Float16)x1[e];
    A1[e] = (_Float16)x2[e]; A1[8 + e] = (_Float16)x3[e];
  }

  _Float16* hl = h1s[wv];
#pragma unroll
  for (int nt = 0; nt < 16; ++nt) {
    v8f acc = {};
    const _Float16* w1row = w1t + (size_t)(nt * 16 + l16) * NC;
    v16h B0 = *(const v16h*)(w1row + 16 * h);
    v16h B1 = *(const v16h*)(w1row + 32 + 16 * h);
    acc = wmma16(A0, B0, acc);
    acc = wmma16(A1, B1, acc);
    float bb = b1[nt * 16 + l16];
#pragma unroll
    for (int r = 0; r < 8; ++r) {
      float v = acc[r] + bb;
      v = v > 0.f ? v : 0.f;
      hl[(r + 8 * h) * NP + nt * 16 + l16] = (_Float16)v;
    }
  }
  __syncthreads();

  // GEMM2 A operands from LDS (8 K-chunks of 32)
  v16h Ah[8];
#pragma unroll
  for (int kc = 0; kc < 8; ++kc) {
    const _Float16* ar = hl + (size_t)l16 * NP + kc * 32;
    v8h a0 = *(const v8h*)(ar + 16 * h);
    v8h a1 = *(const v8h*)(ar + 8 + 16 * h);
    Ah[kc] = cat8(a0, a1);
  }

  v8f t[16];
#pragma unroll
  for (int nt = 0; nt < 16; ++nt) {
    v8f acc = {};
#pragma unroll
    for (int kc = 0; kc < 8; ++kc) {
      const _Float16* w2row =
          w2t + (size_t)(nt * 16 + l16) * NP + kc * 32 + 16 * h;
      v16h Bv = *(const v16h*)w2row;
      acc = wmma16(Ah[kc], Bv, acc);
    }
    float bb = b2[nt * 16 + l16];
#pragma unroll
    for (int r = 0; r < 8; ++r) acc[r] += bb;
    t[nt] = acc;
  }

  // per-row sum of squares -> reduce across 16 lanes -> scale & store
  float ss[8] = {0, 0, 0, 0, 0, 0, 0, 0};
#pragma unroll
  for (int nt = 0; nt < 16; ++nt)
#pragma unroll
    for (int r = 0; r < 8; ++r) ss[r] = fmaf(t[nt][r], t[nt][r], ss[r]);
#pragma unroll
  for (int m = 8; m; m >>= 1)
#pragma unroll
    for (int r = 0; r < 8; ++r) ss[r] += __shfl_xor(ss[r], m, 16);
  float scl[8];
#pragma unroll
  for (int r = 0; r < 8; ++r) {
    float nrm = fmaxf(sqrtf(ss[r]), 1e-12f);
    scl[r] = 1.f / nrm;
  }
#pragma unroll
  for (int nt = 0; nt < 16; ++nt)
#pragma unroll
    for (int r = 0; r < 8; ++r)
      emb[((size_t)b * NK + i0 + r + 8 * h) * NP + nt * 16 + l16] =
          t[nt][r] * scl[r];
}

// ---------------------------------------------------------------------------
extern "C" void kernel_launch(void* const* d_in, const int* in_sizes, int n_in,
                              void* d_out, int out_size, void* d_ws,
                              size_t ws_size, hipStream_t stream) {
  const float* x  = (const float*)d_in[0];  // layer_out [B][C][N]
  const float* w1 = (const float*)d_in[1];  // [C][P]
  const float* b1 = (const float*)d_in[2];  // [P]
  const float* w2 = (const float*)d_in[3];  // [P][P]
  const float* b2 = (const float*)d_in[4];  // [P]

  // workspace carve-out (256B aligned)
  size_t off = 0;
  auto carve = [&](size_t bytes) -> char* {
    char* p = (char*)d_ws + off;
    off = (off + bytes + 255) & ~(size_t)255;
    return p;
  };
  _Float16* pb    = (_Float16*)carve((size_t)NB * NN * NC * 2);
  _Float16* pbT   = (_Float16*)carve((size_t)NB * NC * NN * 2);
  _Float16* w1t   = (_Float16*)carve((size_t)NP * NC * 2);
  _Float16* w2t   = (_Float16*)carve((size_t)NP * NP * 2);
  float*    norm2 = (float*)carve((size_t)NB * NN * 4);
  int*      bmax  = (int*)carve(256);
  float2*   sZ    = (float2*)carve((size_t)NB * NN * 8);
  float*    ent   = (float*)carve((size_t)NB * NN * 4);
  float*    smp   = (float*)carve((size_t)NB * NK * NC * 4);
  (void)in_sizes; (void)n_in; (void)out_size; (void)ws_size;

  // output layout: emb | attn_k | idx  (concatenated flat, return order)
  float* emb_out  = (float*)d_out;
  float* attn_out = emb_out + (size_t)NB * NK * NP;
  int*   idx_out  = (int*)(attn_out + (size_t)NB * NK * NN);

  hipMemsetAsync(bmax, 0, NB * sizeof(int), stream);

  k_prep_rows<<<(NB * NN) / 256, 256, 0, stream>>>(x, pb, norm2, bmax);
  k_cvt<<<(NB * NC * NN) / 256, 256, 0, stream>>>(x, pbT, NB * NC * NN);
  k_wt<<<(NP * NP) / 256, 256, 0, stream>>>(w1, w2, w1t, w2t);

  k_entropy<<<128, 256, 0, stream>>>(pb, norm2, bmax, ent, sZ);  // 1024 waves
  k_topk<<<NB, 256, 0, stream>>>(ent, idx_out);
  k_attn<<<8, 256, 0, stream>>>(pb, sZ, idx_out, attn_out);      // 64 waves
  k_sampled<<<8, 256, 0, stream>>>(attn_out, pbT, smp);          // 64 waves
  k_mlp<<<16, 128, 0, stream>>>(smp, w1t, w2t, b1, b2, emb_out); // 64 waves
}
